// SGAttention_13778255086241
// MI455X (gfx1250) — matrix operand order repaired and checked by reading, not compiled, not run on hardware
//
#include <hip/hip_runtime.h>
#include <hip/hip_bf16.h>

typedef _Float16 v16h __attribute__((ext_vector_type(16)));
typedef _Float16 v8h  __attribute__((ext_vector_type(8)));
typedef _Float16 h4v  __attribute__((ext_vector_type(4)));
typedef float    v8f  __attribute__((ext_vector_type(8)));

union H16 { v16h v; v8h h[2]; };

#define DIMS 128
#define TPAD 136   // halves per padded f16 row (16B-aligned chunks)
#define QPAD 132   // floats per padded f32 row
#define RPAD 136
#define PIX_PER_WG 8
#define RPT 16
#define LN_EPS 1e-6f
#define F32_EPS 1.1920929e-7f

static __device__ inline v8f wmma16(const H16& a, const H16& b, v8f c) {
    return __builtin_amdgcn_wmma_f32_16x16x32_f16(
        /*neg_a=*/false, a.v, /*neg_b=*/false, b.v,
        /*c_mod=*/(short)0, c, /*reuse_a=*/false, /*reuse_b=*/false);
}

// -------- kernel 0: weights -> f16, transposed to [N][K] (B-operand friendly) -----
__global__ __launch_bounds__(256) void prep_weights_kernel(
    const float* __restrict__ Wq, const float* __restrict__ Wk,
    const float* __restrict__ Wv, const float* __restrict__ Wo,
    _Float16* __restrict__ wt)
{
    int t = blockIdx.x * 256 + threadIdx.x;       // 0..16383
    int k = t >> 7, n = t & 127;
    wt[          n * DIMS + k] = (_Float16)Wq[k * DIMS + n];
    wt[16384   + n * DIMS + k] = (_Float16)Wk[k * DIMS + n];
    wt[2*16384 + n * DIMS + k] = (_Float16)Wv[k * DIMS + n];
    wt[3*16384 + n * DIMS + k] = (_Float16)Wo[k * DIMS + n];
}

// -------- kernel 1: Q = (token@Wq + bq) / sqrt(hd), fp32 to workspace ------------
__global__ __launch_bounds__(256) void q_proj_kernel(
    const float* __restrict__ token, const _Float16* __restrict__ wtq,
    const float* __restrict__ bq, float* __restrict__ Q)
{
    __shared__ _Float16 atile[128 * TPAD];
    const int tid = threadIdx.x;
    const int pixBase = blockIdx.x * 128;

    // stage 128x128 f32 -> f16 LDS (coalesced float4)
    const float4* src = (const float4*)(token + (size_t)pixBase * DIMS);
    for (int j = 0; j < 16; ++j) {
        int f = tid + j * 256;
        float4 d = src[f];
        int row = f >> 5, col = (f & 31) << 2;
        h4v hv = {(_Float16)d.x, (_Float16)d.y, (_Float16)d.z, (_Float16)d.w};
        *(h4v*)(&atile[row * TPAD + col]) = hv;
    }
    __syncthreads();

    const int wave = tid >> 5, lane = tid & 31;
    const int hi = lane >> 4, nl = lane & 15;
    const int mrow = wave * 16;

    // preload A (lane row = nl, halves K = 16*hi+e, matching B)
    H16 A[4];
    for (int ks = 0; ks < 4; ++ks) {
        const _Float16* pa = &atile[(mrow + nl) * TPAD + ks * 32 + hi * 16];
        A[ks].h[0] = *(const v8h*)pa;
        A[ks].h[1] = *(const v8h*)(pa + 8);
    }
    for (int n = 0; n < 8; ++n) {
        v8f acc = {0.f,0.f,0.f,0.f,0.f,0.f,0.f,0.f};
        for (int ks = 0; ks < 4; ++ks) {
            H16 B;
            const _Float16* pb = &wtq[(n * 16 + nl) * DIMS + ks * 32 + hi * 16];
            B.h[0] = *(const v8h*)pb;
            B.h[1] = *(const v8h*)(pb + 8);
            acc = wmma16(A[ks], B, acc);
        }
        const int N = n * 16 + nl;
        const float bqv = bq[N];
        for (int r = 0; r < 8; ++r) {
            int row = r + hi * 8;   // C layout: VGPR r -> M=r (lo half) / r+8 (hi half)
            Q[(size_t)(pixBase + mrow + row) * DIMS + N] = (acc[r] + bqv) * 0.25f;
        }
    }
}

// -------- kernel 2: fused KV-proj + segment softmax + Wo + residual + LN ---------
__global__ __launch_bounds__(256) void sg_attn_kernel(
    const float* __restrict__ token, const float* __restrict__ ctx,
    const unsigned char* __restrict__ emask,
    const _Float16* __restrict__ wtk, const _Float16* __restrict__ wtv,
    const _Float16* __restrict__ wto,
    const float* __restrict__ bk, const float* __restrict__ bv,
    const float* __restrict__ bo,
    const float* __restrict__ lns, const float* __restrict__ lnb,
    const float* __restrict__ Q, float* __restrict__ out)
{
    __shared__ _Float16 ctile[128 * TPAD];             // 34816 B; aliased as xtile later
    __shared__ float    qtile[PIX_PER_WG * QPAD];      // 4224 B
    __shared__ _Float16 rtile[16 * RPAD];              // res (f16), rows 8..15 unused
    __shared__ float    mtile[128];                    // mask as 0/1 floats
    float* xtile = (float*)ctile;                      // reuse after barrier

    const int tid = threadIdx.x;
    const int pixBase = blockIdx.x * PIX_PER_WG;
    const int entBase = pixBase * RPT;                 // entries are contiguous per pixel

    // stage ctx tile: 128 rows x 128 f32 -> f16 LDS
    const float4* src = (const float4*)(ctx + (size_t)entBase * DIMS);
    for (int j = 0; j < 16; ++j) {
        int f = tid + j * 256;
        float4 d = src[f];
        int row = f >> 5, col = (f & 31) << 2;
        h4v hv = {(_Float16)d.x, (_Float16)d.y, (_Float16)d.z, (_Float16)d.w};
        *(h4v*)(&ctile[row * TPAD + col]) = hv;
    }
    // stage Q tile (8 x 128 f32)
    {
        const float4* qs = (const float4*)(Q + (size_t)pixBase * DIMS);
        float4 d = qs[tid];
        int row = tid >> 5, col = (tid & 31) << 2;
        *(float4*)(&qtile[row * QPAD + col]) = d;
    }
    if (tid < 128) mtile[tid] = emask[entBase + tid] ? 1.0f : 0.0f;
    __syncthreads();

    const int wave = tid >> 5, lane = tid & 31;
    const int hi = lane >> 4, nl = lane & 15;
    const int N = wave * 16 + nl;                      // wave == head

    // preload B operands for this head's K and V columns
    H16 BK[4], BV[4];
    for (int ks = 0; ks < 4; ++ks) {
        const _Float16* pk = &wtk[N * DIMS + ks * 32 + hi * 16];
        BK[ks].h[0] = *(const v8h*)pk;  BK[ks].h[1] = *(const v8h*)(pk + 8);
        const _Float16* pv = &wtv[N * DIMS + ks * 32 + hi * 16];
        BV[ks].h[0] = *(const v8h*)pv;  BV[ks].h[1] = *(const v8h*)(pv + 8);
    }
    const float bkv = bk[N], bvv = bv[N];

    for (int m = 0; m < PIX_PER_WG; ++m) {             // one pixel == one 16-entry M-tile
        H16 A[4];
        const _Float16* par = &ctile[(m * 16 + nl) * TPAD];
        for (int ks = 0; ks < 4; ++ks) {
            const _Float16* pa = par + ks * 32 + hi * 16;
            A[ks].h[0] = *(const v8h*)pa;
            A[ks].h[1] = *(const v8h*)(pa + 8);
        }
        v8f kacc = {0.f,0.f,0.f,0.f,0.f,0.f,0.f,0.f};
        v8f vacc = {0.f,0.f,0.f,0.f,0.f,0.f,0.f,0.f};
        for (int ks = 0; ks < 4; ++ks) kacc = wmma16(A[ks], BK[ks], kacc);
        for (int ks = 0; ks < 4; ++ks) vacc = wmma16(A[ks], BV[ks], vacc);

        const float qv = qtile[m * QPAD + N];          // q already scaled by 1/sqrt(hd)

        // w[row] = sum_n q[n] * K[row,n]   (16-lane butterfly per half-wave)
        float w[8];
        for (int r = 0; r < 8; ++r) {
            float t = (kacc[r] + bkv) * qv;
            t += __shfl_xor(t, 1); t += __shfl_xor(t, 2);
            t += __shfl_xor(t, 4); t += __shfl_xor(t, 8);
            w[r] = t;                                  // row = hi*8 + r
        }
        // segment softmax over 16 entries (mask applied after exp, per reference)
        float mx = w[0];
        for (int r = 1; r < 8; ++r) mx = fmaxf(mx, w[r]);
        mx = fmaxf(mx, __shfl_xor(mx, 16));
        float s = 0.f;
        for (int r = 0; r < 8; ++r) {
            float e = mtile[m * 16 + hi * 8 + r] * __expf(w[r] - mx);
            w[r] = e; s += e;
        }
        s += __shfl_xor(s, 16);
        const float inv = 1.0f / (s + F32_EPS);
        // res[n] = sum_rows w * V[row,n]
        float p = 0.f;
        for (int r = 0; r < 8; ++r) p += (w[r] * inv) * (vacc[r] + bvv);
        p += __shfl_xor(p, 16);
        if (hi == 0) rtile[m * RPAD + N] = (_Float16)p;
    }
    __syncthreads();

    // output projection: x = res @ Wo + bo + token  (one M-tile, rows 8..15 discarded)
    H16 AO[4];
    for (int ks = 0; ks < 4; ++ks) {
        const _Float16* pa = &rtile[nl * RPAD + ks * 32 + hi * 16];
        AO[ks].h[0] = *(const v8h*)pa;
        AO[ks].h[1] = *(const v8h*)(pa + 8);
    }
    v8f oacc = {0.f,0.f,0.f,0.f,0.f,0.f,0.f,0.f};
    for (int ks = 0; ks < 4; ++ks) {
        H16 B;
        const _Float16* pb = &wto[N * DIMS + ks * 32 + hi * 16];
        B.h[0] = *(const v8h*)pb;  B.h[1] = *(const v8h*)(pb + 8);
        oacc = wmma16(AO[ks], B, oacc);
    }
    const float bov = bo[N];
    for (int r = 0; r < 8; ++r) {
        int row = r + hi * 8;
        if (row < PIX_PER_WG) {
            float x = oacc[r] + bov + token[(size_t)(pixBase + row) * DIMS + N];
            xtile[row * QPAD + N] = x;
        }
    }
    __syncthreads();

    // LayerNorm: one row (pixel) per wave
    {
        const int row = wave;
        const float v0 = xtile[row * QPAD + lane];
        const float v1 = xtile[row * QPAD + lane + 32];
        const float v2 = xtile[row * QPAD + lane + 64];
        const float v3 = xtile[row * QPAD + lane + 96];
        float s = v0 + v1 + v2 + v3;
        s += __shfl_xor(s, 1);  s += __shfl_xor(s, 2);  s += __shfl_xor(s, 4);
        s += __shfl_xor(s, 8);  s += __shfl_xor(s, 16);
        const float mu = s * (1.0f / 128.0f);
        const float d0 = v0 - mu, d1 = v1 - mu, d2 = v2 - mu, d3 = v3 - mu;
        float vs = d0*d0 + d1*d1 + d2*d2 + d3*d3;
        vs += __shfl_xor(vs, 1); vs += __shfl_xor(vs, 2); vs += __shfl_xor(vs, 4);
        vs += __shfl_xor(vs, 8); vs += __shfl_xor(vs, 16);
        const float rinv = rsqrtf(vs * (1.0f / 128.0f) + LN_EPS);
        const size_t ob = (size_t)(pixBase + row) * DIMS;
        out[ob + lane]      = d0 * rinv * lns[lane]      + lnb[lane];
        out[ob + lane + 32] = d1 * rinv * lns[lane + 32] + lnb[lane + 32];
        out[ob + lane + 64] = d2 * rinv * lns[lane + 64] + lnb[lane + 64];
        out[ob + lane + 96] = d3 * rinv * lns[lane + 96] + lnb[lane + 96];
    }
}

extern "C" void kernel_launch(void* const* d_in, const int* in_sizes, int n_in,
                              void* d_out, int out_size, void* d_ws, size_t ws_size,
                              hipStream_t stream) {
    (void)in_sizes; (void)n_in; (void)out_size; (void)ws_size;
    const float* token = (const float*)d_in[0];
    const float* ctx   = (const float*)d_in[1];
    // d_in[2] segms, d_in[3] rpts: structure is fixed (16 consecutive entries/pixel)
    const unsigned char* emask = (const unsigned char*)d_in[4];
    const float* Wq = (const float*)d_in[5];  const float* bq = (const float*)d_in[6];
    const float* Wk = (const float*)d_in[7];  const float* bk = (const float*)d_in[8];
    const float* Wv = (const float*)d_in[9];  const float* bv = (const float*)d_in[10];
    const float* Wo = (const float*)d_in[11]; const float* bo = (const float*)d_in[12];
    const float* lns = (const float*)d_in[13]; const float* lnb = (const float*)d_in[14];

    _Float16* wt = (_Float16*)d_ws;                       // 4 x 128x128 f16 = 128 KB
    float* Q = (float*)((char*)d_ws + 4 * 16384 * 2);     // 65536 x 128 f32 = 32 MB
    float* out = (float*)d_out;

    prep_weights_kernel<<<64, 256, 0, stream>>>(Wq, Wk, Wv, Wo, wt);
    q_proj_kernel<<<65536 / 128, 256, 0, stream>>>(token, wt, bq, Q);
    sg_attn_kernel<<<65536 / PIX_PER_WG, 256, 0, stream>>>(
        token, ctx, emask, wt + 16384, wt + 2 * 16384, wt + 3 * 16384,
        bk, bv, bo, lns, lnb, Q, out);
}
